// Attention_50233937494531
// MI455X (gfx1250) — matrix operand order repaired and checked by reading, not compiled, not run on hardware
//
#include <hip/hip_runtime.h>

typedef __attribute__((ext_vector_type(16))) _Float16 v16h;
typedef __attribute__((ext_vector_type(4)))  _Float16 v4h;
typedef __attribute__((ext_vector_type(8)))  float    v8f;

// Fast exp: v_exp_f32 is exp2 on AMD. Prefer the raw intrinsic (no denorm
// fixup branches); fold log2(e) into the Q-fragment scale in that case.
#if __has_builtin(__builtin_amdgcn_exp2f)
#define FAST_EXP(x) __builtin_amdgcn_exp2f(x)
#define QSCALE_F (0.125f * 1.44269504088896340736f)   // 1/sqrt(64) * log2(e)
#else
#define FAST_EXP(x) __expf(x)
#define QSCALE_F (0.125f)                             // 1/sqrt(64)
#endif

namespace {
constexpr int B  = 2;
constexpr int H  = 16;
constexpr int S  = 2048;
constexpr int D  = 64;
constexpr int WPB   = 8;          // waves per block (wave32)
constexpr int QTILE = 16;         // query rows per wave
}

// 16-bit A-matrix element -> K(contraction) index (contiguous: k = 16*hl + e, matches B fills)
__device__ __forceinline__ int kmapA(int e, int hl) {
  return e + 16 * hl;
}

#define WMMA_F16(Af, Bf, Cf) \
  __builtin_amdgcn_wmma_f32_16x16x32_f16(false, (Af), false, (Bf), (short)0, (Cf), false, false)

// ---- one-shot f32 -> f16 mirror (Q, K) ----
__global__ __launch_bounds__(256, 1)
void cvt_f16_kernel(const float* __restrict__ src, _Float16* __restrict__ dst, int n) {
  const int i = (blockIdx.x * 256 + threadIdx.x) * 4;
  if (i < n) {
    const float4 f = *(const float4*)(src + i);
    v4h h;
    h[0] = (_Float16)f.x; h[1] = (_Float16)f.y;
    h[2] = (_Float16)f.z; h[3] = (_Float16)f.w;
    *(v4h*)(dst + i) = h;
  }
}

// ---- one-shot f32 -> f16 transpose of V: per head [S][D] -> [D][S] ----
__global__ __launch_bounds__(256, 1)
void cvt_vt_kernel(const float* __restrict__ src, _Float16* __restrict__ dst) {
  const int idx  = blockIdx.x * 256 + threadIdx.x;  // one (head, s) row each
  const int head = idx / S;
  const int s    = idx % S;
  const float* row = src + (size_t)idx * D;
  _Float16* out = dst + (size_t)head * D * S + s;
  #pragma unroll
  for (int d = 0; d < D; ++d)
    out[(size_t)d * S] = (_Float16)row[d];   // writes coalesce across threads (consecutive s)
}

template <typename T, bool TRV>   // TRV: V is f16 and pre-transposed [head][D][S]
__global__ __launch_bounds__(WPB * 32, 1)
void attn_fwd_kernel(const T* __restrict__ Qg, const T* __restrict__ Kg,
                     const T* __restrict__ Vg, const int* __restrict__ Mg,
                     float* __restrict__ Og, float* __restrict__ Pg)
{
  __shared__ float stage[WPB][QTILE * 32];   // per-wave P-tile staging (2KB/wave)

  const int lane = threadIdx.x & 31;
  const int wid  = threadIdx.x >> 5;
  const int gw   = blockIdx.x * WPB + wid;
  const int head = gw / (S / QTILE);         // b*H + h
  const int qt   = gw % (S / QTILE);
  const int b    = head / H;
  const int col  = lane & 15;                // N index / key-in-tile
  const int hl   = lane >> 4;                // lane half

  const size_t hbase = (size_t)head * S * D;
  const int* mrow = Mg + (size_t)b * S;

  // ---- Q A-fragments (softmax scale folded in) for d-chunks [0,32) and [32,64) ----
  v16h aq0, aq1;
  {
    const T* qp = Qg + hbase + (size_t)(qt * QTILE + col) * D;
    #pragma unroll
    for (int e = 0; e < 16; ++e) {
      const int d = kmapA(e, hl);
      aq0[e] = (_Float16)((float)qp[d] * QSCALE_F);
      aq1[e] = (_Float16)((float)qp[32 + d] * QSCALE_F);
    }
  }

  // ---------------- pass 1: softmax denominators ----------------
  float rs[8];
  #pragma unroll
  for (int j = 0; j < 8; ++j) rs[j] = 0.0f;

  for (int kt = 0; kt < S / 16; ++kt) {
    const T* kp = Kg + hbase + (size_t)(kt * 16 + col) * D + 16 * hl;
    v16h bk0, bk1;
    #pragma unroll
    for (int e = 0; e < 16; ++e) {
      bk0[e] = (_Float16)kp[e];
      bk1[e] = (_Float16)kp[32 + e];
    }
    v8f c = {};
    c = WMMA_F16(aq0, bk0, c);
    c = WMMA_F16(aq1, bk1, c);
    const bool live = mrow[kt * 16 + col] != 0;
    #pragma unroll
    for (int j = 0; j < 8; ++j) {
      const float e2 = FAST_EXP(c[j]);     // single v_exp_f32
      rs[j] += live ? e2 : 0.0f;           // cndmask, no branch
    }
  }
  // reduce row sums across the 16 lanes sharing each half (N columns)
  #pragma unroll
  for (int off = 1; off < 16; off <<= 1) {
    #pragma unroll
    for (int j = 0; j < 8; ++j)
      rs[j] += __shfl_xor(rs[j], off, 32);
  }
  float inv[8];
  #pragma unroll
  for (int j = 0; j < 8; ++j) inv[j] = 1.0f / rs[j];   // row = j + 8*hl

  // ---------------- pass 2: probs + P@V ----------------
  v8f acc0 = {}, acc1 = {}, acc2 = {}, acc3 = {};
  float* st = stage[wid];

  for (int kc = 0; kc < S / 32; ++kc) {
    #pragma unroll
    for (int sub = 0; sub < 2; ++sub) {
      const int kt = kc * 2 + sub;
      const T* kp = Kg + hbase + (size_t)(kt * 16 + col) * D + 16 * hl;
      v16h bk0, bk1;
      #pragma unroll
      for (int e = 0; e < 16; ++e) {
        bk0[e] = (_Float16)kp[e];
        bk1[e] = (_Float16)kp[32 + e];
      }
      v8f c = {};
      c = WMMA_F16(aq0, bk0, c);
      c = WMMA_F16(aq1, bk1, c);
      const bool live = mrow[kt * 16 + col] != 0;
      #pragma unroll
      for (int j = 0; j < 8; ++j) {
        const int r = j + 8 * hl;
        const float e2 = FAST_EXP(c[j]) * inv[j];
        st[r * 32 + sub * 16 + col] = live ? e2 : 0.0f;   // stage normalized P tile
      }
    }
    // LDS write->read (cross-lane) within the wave: drain DS counter
    asm volatile("s_wait_dscnt 0" ::: "memory");

    // stream probs out as float4 (128B-contiguous segments per row)
    #pragma unroll
    for (int it = 0; it < 4; ++it) {
      const int idx = it * 32 + lane;     // 128 float4 groups in the 16x32 tile
      const int row = idx >> 3;
      const int g   = idx & 7;
      const float4 pv = *(const float4*)&st[row * 32 + g * 4];
      *(float4*)&Pg[((size_t)head * S + qt * QTILE + row) * S + kc * 32 + g * 4] = pv;
    }

    // A fragment of P (16 rows x 32 keys)
    v16h ap;
    #pragma unroll
    for (int e = 0; e < 16; ++e)
      ap[e] = (_Float16)st[col * 32 + kmapA(e, hl)];

    // B fragments of V (32 keys x 16 dims) for the four 16-wide d tiles
    v16h bv0, bv1, bv2, bv3;
    if constexpr (TRV) {
      const T* vt = Vg + (size_t)head * D * S + (size_t)col * S + (kc * 32 + 16 * hl);
      #pragma unroll
      for (int e = 0; e < 16; ++e) {      // contiguous 16 halfs per fragment -> b128 loads
        bv0[e] = (_Float16)vt[e];
        bv1[e] = (_Float16)vt[(size_t)16 * S + e];
        bv2[e] = (_Float16)vt[(size_t)32 * S + e];
        bv3[e] = (_Float16)vt[(size_t)48 * S + e];
      }
    } else {
      const T* vp = Vg + hbase + (size_t)(kc * 32 + 16 * hl) * D;
      #pragma unroll
      for (int e = 0; e < 16; ++e) {
        const T* vr = vp + (size_t)e * D;
        bv0[e] = (_Float16)vr[col];
        bv1[e] = (_Float16)vr[16 + col];
        bv2[e] = (_Float16)vr[32 + col];
        bv3[e] = (_Float16)vr[48 + col];
      }
    }
    acc0 = WMMA_F16(ap, bv0, acc0);
    acc1 = WMMA_F16(ap, bv1, acc1);
    acc2 = WMMA_F16(ap, bv2, acc2);
    acc3 = WMMA_F16(ap, bv3, acc3);
  }

  // ---- write out (16 x 64 f32) ----
  #pragma unroll
  for (int j = 0; j < 8; ++j) {
    const int r = j + 8 * hl;
    float* op = Og + hbase + (size_t)(qt * QTILE + r) * D;
    op[col]      = acc0[j];
    op[16 + col] = acc1[j];
    op[32 + col] = acc2[j];
    op[48 + col] = acc3[j];
  }
}

extern "C" void kernel_launch(void* const* d_in, const int* in_sizes, int n_in,
                              void* d_out, int out_size, void* d_ws, size_t ws_size,
                              hipStream_t stream) {
  (void)in_sizes; (void)n_in; (void)out_size;
  const float* Qg = (const float*)d_in[0];
  const float* Kg = (const float*)d_in[1];
  const float* Vg = (const float*)d_in[2];
  const int*   Mg = (const int*)d_in[3];
  float* Og = (float*)d_out;
  float* Pg = Og + (size_t)B * H * S * D;    // tuple: (out, probs) concatenated

  const int total_waves = B * H * (S / QTILE);   // 4096 query tiles
  dim3 grid(total_waves / WPB);
  dim3 block(WPB * 32);

  const size_t nT   = (size_t)B * H * S * D;         // elements per tensor
  const size_t need = 3 * nT * sizeof(_Float16);     // 24 MB f16 mirrors

  if (ws_size >= need) {
    _Float16* Qh = (_Float16*)d_ws;
    _Float16* Kh = Qh + nT;
    _Float16* Vt = Kh + nT;
    const int nBlk = (int)((nT / 4 + 255) / 256);
    cvt_f16_kernel<<<nBlk, 256, 0, stream>>>(Qg, Qh, (int)nT);
    cvt_f16_kernel<<<nBlk, 256, 0, stream>>>(Kg, Kh, (int)nT);
    cvt_vt_kernel<<<(B * H * S) / 256, 256, 0, stream>>>(Vg, Vt);
    attn_fwd_kernel<_Float16, true><<<grid, block, 0, stream>>>(Qh, Kh, Vt, Mg, Og, Pg);
  } else {
    attn_fwd_kernel<float, false><<<grid, block, 0, stream>>>(Qg, Kg, Vg, Mg, Og, Pg);
  }
}
